// Generator_17832704213029
// MI455X (gfx1250) — matrix operand order repaired and checked by reading, not compiled, not run on hardware
//
#include <hip/hip_runtime.h>

typedef __attribute__((ext_vector_type(16))) _Float16 v16h;
typedef __attribute__((ext_vector_type(8)))  _Float16 v8h;
typedef __attribute__((ext_vector_type(8)))  float    v8f;

#define NEG_SLOPE 0.2f

__device__ __forceinline__ void atomicMaxF(float* addr, float val) {
  if (val >= 0.f) atomicMax((int*)addr, __float_as_int(val));
  else            atomicMin((unsigned int*)addr, __float_as_uint(val));
}

__device__ __forceinline__ int imin(int a, int b) { return a < b ? a : b; }

__device__ __forceinline__ v16h cat16(v8h lo, v8h hi) {
  return __builtin_shufflevector(lo, hi, 0, 1, 2, 3, 4, 5, 6, 7,
                                 8, 9, 10, 11, 12, 13, 14, 15);
}

__global__ void fill_kernel(float* p, float v, long long n) {
  long long i = (long long)blockIdx.x * blockDim.x + threadIdx.x;
  if (i < n) p[i] = v;
}

__global__ void fill_h_kernel(_Float16* p, long long n) {
  long long i = (long long)blockIdx.x * blockDim.x + threadIdx.x;
  if (i < n) p[i] = (_Float16)0.f;
}

__global__ void scale_kernel(float* p, float s, int n) {
  int i = blockIdx.x * blockDim.x + threadIdx.x;
  if (i < n) p[i] *= s;
}

// Pack f32 [Rows,K] row-major -> fp16 [RowsPad,Kpad], zero-padded.
// Wave-per-row: no integer division, coalesced lanes across K.
__global__ void pack_f16_kernel(const float* __restrict__ S, _Float16* __restrict__ D,
                                const float* __restrict__ zbuf,
                                int Rows, int K, int RowsPad, int Kpad) {
  int r = (int)((blockIdx.x * (long long)blockDim.x + threadIdx.x) >> 5);
  int lane = threadIdx.x & 31;
  if (r >= RowsPad) return;
  const float* Srow = S + (long long)r * K;
  _Float16* Drow = D + (long long)r * Kpad;
  for (int k = lane; k < Kpad; k += 32) {
    const float* p = (r < Rows && k < K) ? (Srow + k) : zbuf;
    Drow[k] = (_Float16)(*p);
  }
}

// Pack extended edge attrs (edges then mean-filled self loops) -> fp16 [EtotPad,32].
__global__ void pack_ea_kernel(const float* __restrict__ ea, const float* __restrict__ eamean,
                               _Float16* __restrict__ D, const float* __restrict__ zbuf,
                               int E, int Etot, long long total, int Ka) {
  long long i = (long long)blockIdx.x * blockDim.x + threadIdx.x;
  if (i >= total) return;
  int e = (int)(i >> 5), k = (int)(i & 31);
  const float* src = (e < E) ? (ea + (long long)e * Ka + k)
                             : (eamean + (long long)(e - E) * Ka + k);
  const float* p = (e < Etot && k < Ka) ? src : zbuf;
  D[i] = (_Float16)(*p);
}

// ---------------------------------------------------------------------------
// GEMM on pre-packed fp16 operands:
//   C[M,N] = act( A[M,K] @ B[Npad,K]^T + bias[N] )
// A: fp16 [M, lda]  (lda = Kpad, mult of 32; M mult of 16; zero padded)
// B: fp16 [Npad, ldb] (weights, zero padded)
// Outputs: optional f32 C (stride N) and/or fp16 C16 (stride ld16).
// Inner loop: four 16B v8h loads + one v_wmma_f32_16x16x32_f16. No guards.
// ---------------------------------------------------------------------------
__global__ void gemm_f16_kernel(const _Float16* __restrict__ A, int lda,
                                const _Float16* __restrict__ B, int ldb,
                                const float* __restrict__ bias,
                                float* __restrict__ C, _Float16* __restrict__ C16, int ld16,
                                const float* __restrict__ zbuf,
                                int M, int N, int Npad, int K, int act) {
  int wave = (int)((blockIdx.x * (long long)blockDim.x + threadIdx.x) >> 5);
  int lane = threadIdx.x & 31;
  int tilesN = Npad >> 4;
  int tilesM = M >> 4;
  if (wave >= tilesM * tilesN) return;            // wave-uniform
  int tm = wave / tilesN, tn = wave % tilesN;
  int half = lane >> 4;
  int rowA = tm * 16 + (lane & 15);
  int colB = tn * 16 + (lane & 15);
  const _Float16* __restrict__ Ab = A + (long long)rowA * lda;
  const _Float16* __restrict__ Bb = B + (long long)colB * ldb;

  v8f acc = {};
  for (int k0 = 0; k0 < K; k0 += 32) {
    v8h a0 = *(const v8h*)(Ab + k0 + half * 8);        // K 0..7 / 8..15
    v8h a1 = *(const v8h*)(Ab + k0 + 16 + half * 8);   // K 16..23 / 24..31
    v8h b0 = *(const v8h*)(Bb + k0 + half * 8);
    v8h b1 = *(const v8h*)(Bb + k0 + 16 + half * 8);
    acc = __builtin_amdgcn_wmma_f32_16x16x32_f16(false, cat16(a0, a1),
                                                 false, cat16(b0, b1),
                                                 (short)0, acc, false, false);
  }
  int c = colB;
  bool cOk = c < N;
  const float* bp = cOk ? (bias + c) : zbuf;           // single hoisted bias load
  float bc = *bp;
#pragma unroll
  for (int g = 0; g < 8; ++g) {
    int r = tm * 16 + g + (half << 3);
    float v = acc[g] + bc;
    if (act) v = fmaxf(v, 0.f);
    if (cOk) {
      if (C)   C[(long long)r * N + c] = v;
      if (C16) C16[(long long)r * ld16 + c] = (_Float16)v;
    }
  }
}

// ---------------------------------------------------------------------------
// Per-edge degree + edge_attr sum for fill_value='mean' self-loops.
// Wave-per-edge: lane 0 counts degree, lanes < Ka scatter the attrs.
// ---------------------------------------------------------------------------
__global__ void deg_attr_kernel(const int* __restrict__ ei, const float* __restrict__ ea,
                                float* deg, float* easum, int E, int Ka) {
  int e = (int)((blockIdx.x * (long long)blockDim.x + threadIdx.x) >> 5);
  int lane = threadIdx.x & 31;
  if (e >= E) return;                              // wave-uniform
  int d = ei[(long long)E + e];
  if (lane == 0) atomicAdd(&deg[d], 1.0f);
  if (lane < Ka)
    atomicAdd(&easum[(long long)d * Ka + lane], ea[(long long)e * Ka + lane]);
}

// Wave-per-node: easum[n][:] /= max(deg[n],1)
__global__ void mean_attr_kernel(float* easum, const float* deg, int Nn, int Ka) {
  int n = (int)((blockIdx.x * (long long)blockDim.x + threadIdx.x) >> 5);
  int lane = threadIdx.x & 31;
  if (n >= Nn) return;
  float inv = 1.0f / fmaxf(deg[n], 1.0f);
  if (lane < Ka) easum[(long long)n * Ka + lane] *= inv;
}

// ---------------------------------------------------------------------------
// Fused GATv2 edge pass on packed fp16 operands. One wave owns 16 extended
// edges. eW tile via WMMA (A = eah [EtotPad,32], B = Weh [Fpad,32]), fused
// fp16 gathers of xl[src]/xr[dst], leaky-ReLU, dot with att, butterfly
// reduce across each 16-lane half, write logits + atomic-max m[dst].
// Fragment loads are fully unguarded (padded & zero-filled buffers).
// ---------------------------------------------------------------------------
__global__ void edge_logits_kernel(const int* __restrict__ ei,
                                   const _Float16* __restrict__ eah,
                                   const _Float16* __restrict__ xlh,
                                   const _Float16* __restrict__ xrh,
                                   const _Float16* __restrict__ Weh,
                                   const float* __restrict__ att,
                                   float* __restrict__ logits, float* __restrict__ mbuf,
                                   const float* __restrict__ zbuf,
                                   int E, int Nn, int F, int Fpad) {
  int wave = (int)((blockIdx.x * (long long)blockDim.x + threadIdx.x) >> 5);
  int lane = threadIdx.x & 31;
  int Etot = E + Nn;
  int ebase = wave * 16;
  if (ebase >= Etot) return;                      // wave-uniform
  int half = lane >> 4;

  // A fragment: 16 packed edge-attr rows (K padded to 32, zero filled)
  int ra = ebase + (lane & 15);
  const _Float16* __restrict__ Ab = eah + (long long)ra * 32;
  v8h a0 = *(const v8h*)(Ab + half * 8);
  v8h a1 = *(const v8h*)(Ab + 16 + half * 8);
  v16h afrag = cat16(a0, a1);

  // src/dst for this lane's accumulator rows r = g + 8*half (0 when invalid)
  int sArr[8], dArr[8];
#pragma unroll
  for (int g = 0; g < 8; ++g) {
    int er = ebase + g + 8 * half;
    bool ok = er < Etot;
    int erC = imin(er, Etot - 1);
    int s, d;
    if (erC < E) { s = ei[erC]; d = ei[(long long)E + erC]; }
    else         { s = erC - E; d = erC - E; }
    sArr[g] = ok ? s : 0;
    dArr[g] = ok ? d : 0;
  }

  float partial[8] = {0.f, 0.f, 0.f, 0.f, 0.f, 0.f, 0.f, 0.f};
  int tilesF = Fpad >> 4;
  for (int t = 0; t < tilesF; ++t) {
    int cB = t * 16 + (lane & 15);                // < Fpad always
    const _Float16* __restrict__ Bb = Weh + (long long)cB * 32;
    v8h b0 = *(const v8h*)(Bb + half * 8);
    v8h b1 = *(const v8h*)(Bb + 16 + half * 8);
    v8f acc = {};
    acc = __builtin_amdgcn_wmma_f32_16x16x32_f16(false, afrag, false, cat16(b0, b1),
                                                 (short)0, acc, false, false);
    bool cOk = cB < F;
    int cC = imin(cB, F - 1);
    const float* attp = cOk ? (att + cB) : zbuf;  // 0 kills padded columns
    float attc = *attp;
#pragma unroll
    for (int g = 0; g < 8; ++g) {
      float h = (float)xlh[(long long)sArr[g] * F + cC] +
                (float)xrh[(long long)dArr[g] * F + cC] + acc[g];
      h = (h > 0.f) ? h : h * NEG_SLOPE;
      partial[g] += attc * h;
    }
  }

  // reduce over the 16 lanes of each half (xor masks < 16 keep halves separate)
#pragma unroll
  for (int g = 0; g < 8; ++g) {
    float v = partial[g];
    v += __shfl_xor(v, 1, 32);
    v += __shfl_xor(v, 2, 32);
    v += __shfl_xor(v, 4, 32);
    v += __shfl_xor(v, 8, 32);
    partial[g] = v;
  }

  int gsel = lane & 7;
  float val = 0.f; int dsel = 0;
  switch (gsel) {
    case 0: val = partial[0]; dsel = dArr[0]; break;
    case 1: val = partial[1]; dsel = dArr[1]; break;
    case 2: val = partial[2]; dsel = dArr[2]; break;
    case 3: val = partial[3]; dsel = dArr[3]; break;
    case 4: val = partial[4]; dsel = dArr[4]; break;
    case 5: val = partial[5]; dsel = dArr[5]; break;
    case 6: val = partial[6]; dsel = dArr[6]; break;
    default: val = partial[7]; dsel = dArr[7]; break;
  }
  if ((lane & 15) < 8) {
    int er = ebase + gsel + 8 * half;
    if (er < Etot) {
      logits[er] = val;
      atomicMaxF(&mbuf[dsel], val);
    }
  }
}

__global__ void softmax_p_kernel(const int* __restrict__ ei, const float* __restrict__ logits,
                                 const float* __restrict__ mbuf, float* __restrict__ p,
                                 float* __restrict__ denom, int E, int Nn) {
  long long e = (long long)blockIdx.x * blockDim.x + threadIdx.x;
  if (e >= (long long)E + Nn) return;
  int d = (e < E) ? ei[(long long)E + e] : (int)(e - E);
  float pv = __expf(logits[e] - mbuf[d]);
  p[e] = pv;
  atomicAdd(&denom[d], pv);
}

// Wave-per-edge: alpha computed once, lanes stride the feature row.
__global__ void aggregate_kernel(const int* __restrict__ ei, const float* __restrict__ p,
                                 const float* __restrict__ denom,
                                 const _Float16* __restrict__ xlh,
                                 float* __restrict__ out, int E, int Nn, int F) {
  int e = (int)((blockIdx.x * (long long)blockDim.x + threadIdx.x) >> 5);
  int lane = threadIdx.x & 31;
  int Etot = E + Nn;
  if (e >= Etot) return;                           // wave-uniform
  int s, d;
  if (e < E) { s = ei[e]; d = ei[(long long)E + e]; }
  else       { s = e - E; d = s; }
  float alpha = p[e] / denom[d];
  const _Float16* __restrict__ xrow = xlh + (long long)s * F;
  float* orow = out + (long long)d * F;
  for (int f = lane; f < F; f += 32)
    atomicAdd(&orow[f], alpha * (float)xrow[f]);
}

// Wave-per-node bias + optional relu.
__global__ void bias_act_kernel(float* h, const float* __restrict__ b, int Nn, int F, int act) {
  int n = (int)((blockIdx.x * (long long)blockDim.x + threadIdx.x) >> 5);
  int lane = threadIdx.x & 31;
  if (n >= Nn) return;
  float* row = h + (long long)n * F;
  for (int f = lane; f < F; f += 32) {
    float v = row[f] + b[f];
    if (act) v = fmaxf(v, 0.f);
    row[f] = v;
  }
}

// segment mean-pool (rule graphs): wave-per-node
__global__ void pool_seg_kernel(const float* __restrict__ h, const int* __restrict__ seg,
                                float* psum, float* pcnt, int Nn, int F) {
  int n = (int)((blockIdx.x * (long long)blockDim.x + threadIdx.x) >> 5);
  int lane = threadIdx.x & 31;
  if (n >= Nn) return;
  int r = seg[n];
  if (lane == 0) atomicAdd(&pcnt[r], 1.0f);
  const float* hrow = h + (long long)n * F;
  float* prow = psum + (long long)r * F;
  for (int f = lane; f < F; f += 32) atomicAdd(&prow[f], hrow[f]);
}

__global__ void pool_div_kernel(float* psum, const float* pcnt, int R, int F) {
  int i = blockIdx.x * blockDim.x + threadIdx.x;
  if (i >= R * F) return;
  psum[i] /= fmaxf(pcnt[i / F], 1.0f);
}

// whole-graph mean pool (batch == 0): wave-per-node into LDS partial sums,
// then one global atomic per feature per block.
__global__ void pool_all_kernel(const float* __restrict__ g, float* xp, int Nn, int F) {
  __shared__ float sums[256];
  for (int i = threadIdx.x; i < F; i += blockDim.x) sums[i] = 0.f;
  __syncthreads();
  int wv = threadIdx.x >> 5;
  int lane = threadIdx.x & 31;
  int wavesPerBlock = blockDim.x >> 5;
  for (int n = blockIdx.x * wavesPerBlock + wv; n < Nn; n += gridDim.x * wavesPerBlock) {
    const float* grow = g + (long long)n * F;
    for (int f = lane; f < F; f += 32) atomicAdd(&sums[f], grow[f]);   // ds_add_f32
  }
  __syncthreads();
  for (int i = threadIdx.x; i < F; i += blockDim.x) atomicAdd(&xp[i], sums[i]);
}

__global__ void bil_w_kernel(const float* __restrict__ xp, const float* __restrict__ W,
                             float* wvec, int D) {
  int e = blockIdx.x * blockDim.x + threadIdx.x;
  if (e >= D) return;
  float s = 0.f;
  for (int d = 0; d < D; ++d) s += xp[d] * W[d * D + e];
  wvec[e] = s;
}

__global__ void bil_out_kernel(const float* __restrict__ wvec, const float* __restrict__ y,
                               const float* __restrict__ bb, float* out, int R, int D) {
  int r = blockIdx.x * blockDim.x + threadIdx.x;
  if (r >= R) return;
  float s = 0.f;
  for (int e = 0; e < D; ++e) s += wvec[e] * y[(long long)r * D + e];
  out[r] = s + bb[0];
}

// ---------------------------------------------------------------------------

static inline int divup_ll(long long a, int b) { return (int)((a + b - 1) / b); }
static inline int rnd16(int v) { return (v + 15) & ~15; }

extern "C" void kernel_launch(void* const* d_in, const int* in_sizes, int n_in,
                              void* d_out, int out_size, void* d_ws, size_t ws_size,
                              hipStream_t stream) {
  const float* x   = (const float*)d_in[0];
  const float* ea  = (const float*)d_in[1];
  const float* rx  = (const float*)d_in[2];
  const float* rea = (const float*)d_in[3];
  const float* g_[3][7];
  for (int l = 0; l < 3; ++l)
    for (int j = 0; j < 7; ++j) g_[l][j] = (const float*)d_in[4 + l * 7 + j];
  const float* lin_W = (const float*)d_in[25]; const float* lin_b = (const float*)d_in[26];
  const float* f1_W  = (const float*)d_in[27]; const float* f1_b  = (const float*)d_in[28];
  const float* f2_W  = (const float*)d_in[29]; const float* f2_b  = (const float*)d_in[30];
  const float* f3_W  = (const float*)d_in[31]; const float* f3_b  = (const float*)d_in[32];
  const float* bilW  = (const float*)d_in[33]; const float* bilb  = (const float*)d_in[34];
  const int* ei   = (const int*)d_in[35];
  const int* rei  = (const int*)d_in[37];
  const int* rbat = (const int*)d_in[38];

  int N  = in_sizes[0] / 16;     // 80000 (mult of 16)
  int E  = in_sizes[1] / 18;
  int NR = in_sizes[2] / 16;     // 20480 (mult of 16)
  int ER = in_sizes[3] / 18;
  const int R = 512;
  int Etot0 = E + N,  Etot0P = rnd16(Etot0);
  int EtotR = ER + NR, EtotRP = rnd16(EtotR);

  // ---- workspace carve ----
  char* wptr = (char*)d_ws;
  char* wend = (char*)d_ws + ws_size;
  auto allocF = [&](long long nf) -> float* {
    float* p = (float*)wptr;
    wptr += ((nf * 4 + 255) / 256) * 256;
    return p;
  };
  auto allocH = [&](long long nh) -> _Float16* {
    _Float16* p = (_Float16*)wptr;
    wptr += ((nh * 2 + 255) / 256) * 256;
    return p;
  };
  float* zbuf   = allocF(64);
  // f32 scratch
  float* deg0 = allocF(N);            float* eam0 = allocF((long long)N * 18);
  float* log0 = allocF(Etot0);        float* p0   = allocF(Etot0);
  float* m0   = allocF(N);            float* den0 = allocF(N);
  float* gout = allocF((long long)N * 100);
  float* degr = allocF(NR);           float* eamr = allocF((long long)NR * 18);
  float* logr = allocF(EtotR);        float* pr   = allocF(EtotR);
  float* mr   = allocF(NR);           float* denr = allocF(NR);
  float* h1   = allocF((long long)NR * 100);
  float* h2   = allocF((long long)NR * 200);
  float* h3   = allocF((long long)NR * 400);
  float* psum = allocF((long long)R * 400);  float* pcnt = allocF(R);
  float* y3   = allocF((long long)R * 100);
  float* xp   = allocF(128);          float* wvec = allocF(128);
  // packed fp16 operands
  _Float16* xh    = allocH((long long)N * 32);
  _Float16* rxh   = allocH((long long)NR * 32);
  _Float16* eah0  = allocH((long long)Etot0P * 32);
  _Float16* eahr  = allocH((long long)EtotRP * 32);
  _Float16* xl0h  = allocH((long long)N * 100);
  _Float16* xr0h  = allocH((long long)N * 100);
  _Float16* xl1h  = allocH((long long)NR * 100);
  _Float16* xr1h  = allocH((long long)NR * 100);
  _Float16* xl2h  = allocH((long long)NR * 200);
  _Float16* xr2h  = allocH((long long)NR * 200);
  _Float16* h1h   = allocH((long long)NR * 128);
  _Float16* h2h   = allocH((long long)NR * 224);
  _Float16* psumh = allocH((long long)R * 416);
  _Float16* y1h   = allocH((long long)R * 224);
  _Float16* y2h   = allocH((long long)R * 128);
  // packed weights
  _Float16* Wl0h = allocH(112 * 32);  _Float16* Wr0h = allocH(112 * 32);
  _Float16* We0h = allocH(112 * 32);
  _Float16* Wl1h = allocH(112 * 32);  _Float16* Wr1h = allocH(112 * 32);
  _Float16* We1h = allocH(112 * 32);
  _Float16* Wl2h = allocH(208 * 128); _Float16* Wr2h = allocH(208 * 128);
  _Float16* We2h = allocH(208 * 32);
  _Float16* linWh = allocH(400 * 224);
  _Float16* f1Wh  = allocH(208 * 416);
  _Float16* f2Wh  = allocH(112 * 224);
  _Float16* f3Wh  = allocH(112 * 128);
  if (wptr > wend) return;   // workspace too small: bail deterministically

  auto FILL = [&](float* p, float v, long long n) {
    fill_kernel<<<divup_ll(n, 256), 256, 0, stream>>>(p, v, n);
  };
  auto FILLH = [&](_Float16* p, long long n) {
    fill_h_kernel<<<divup_ll(n, 256), 256, 0, stream>>>(p, n);
  };
  auto PACK = [&](const float* S, _Float16* D, int Rows, int K, int RowsPad, int Kpad) {
    pack_f16_kernel<<<divup_ll((long long)RowsPad * 32, 256), 256, 0, stream>>>(
        S, D, zbuf, Rows, K, RowsPad, Kpad);
  };
  auto GEMM = [&](const _Float16* A, int lda, const _Float16* B, int ldb, const float* bias,
                  float* C, _Float16* C16, int ld16, int M, int Nc, int Npad, int K, int act) {
    long long waves = (long long)(M >> 4) * (Npad >> 4);
    gemm_f16_kernel<<<divup_ll(waves * 32, 256), 256, 0, stream>>>(
        A, lda, B, ldb, bias, C, C16, ld16, zbuf, M, Nc, Npad, K, act);
  };

  FILL(zbuf, 0.f, 64);

  // ---- pack all weights ----
  PACK(g_[0][0], Wl0h, 100, 16, 112, 32);  PACK(g_[0][2], Wr0h, 100, 16, 112, 32);
  PACK(g_[0][4], We0h, 100, 18, 112, 32);
  PACK(g_[1][0], Wl1h, 100, 16, 112, 32);  PACK(g_[1][2], Wr1h, 100, 16, 112, 32);
  PACK(g_[1][4], We1h, 100, 18, 112, 32);
  PACK(g_[2][0], Wl2h, 200, 100, 208, 128); PACK(g_[2][2], Wr2h, 200, 100, 208, 128);
  PACK(g_[2][4], We2h, 200, 18, 208, 32);
  PACK(lin_W, linWh, 400, 200, 400, 224);
  PACK(f1_W, f1Wh, 200, 400, 208, 416);
  PACK(f2_W, f2Wh, 100, 200, 112, 224);
  PACK(f3_W, f3Wh, 100, 100, 112, 128);
  PACK(x, xh, N, 16, N, 32);
  PACK(rx, rxh, NR, 16, NR, 32);

  // GATv2 layer driver on packed operands
  auto GAT = [&](const _Float16* featH, int ldaF, const int* ei_, _Float16* eahB,
                 const _Float16* WlH, const float* bl, const _Float16* WrH, const float* br,
                 const _Float16* WeH, const float* att, const float* bb,
                 _Float16* xlh_, _Float16* xrh_, float* lg_, float* p_, float* m_, float* dn_,
                 float* out_, int n_, int e_, int Kp, int fout, int foutPad, int act) {
    GEMM(featH, ldaF, WlH, Kp, bl, nullptr, xlh_, fout, n_, fout, foutPad, Kp, 0);  // lin_l
    GEMM(featH, ldaF, WrH, Kp, br, nullptr, xrh_, fout, n_, fout, foutPad, Kp, 0);  // lin_r
    FILL(m_, -3.4e38f, n_);
    FILL(dn_, 0.f, n_);
    int Etot = e_ + n_;
    long long ewaves = (Etot + 15) / 16;
    edge_logits_kernel<<<divup_ll(ewaves * 32, 256), 256, 0, stream>>>(
        ei_, eahB, xlh_, xrh_, WeH, att, lg_, m_, zbuf, e_, n_, fout, foutPad);
    softmax_p_kernel<<<divup_ll(Etot, 256), 256, 0, stream>>>(ei_, lg_, m_, p_, dn_, e_, n_);
    FILL(out_, 0.f, (long long)n_ * fout);
    aggregate_kernel<<<divup_ll((long long)Etot * 32, 256), 256, 0, stream>>>(
        ei_, p_, dn_, xlh_, out_, e_, n_, fout);
    bias_act_kernel<<<divup_ll((long long)n_ * 32, 256), 256, 0, stream>>>(out_, bb, n_, fout, act);
  };

  // ---- rule graph: degree, mean-attr, packed extended edge attrs (shared) ----
  FILL(degr, 0.f, NR);
  FILL(eamr, 0.f, (long long)NR * 18);
  deg_attr_kernel<<<divup_ll((long long)ER * 32, 256), 256, 0, stream>>>(rei, rea, degr, eamr, ER, 18);
  mean_attr_kernel<<<divup_ll((long long)NR * 32, 256), 256, 0, stream>>>(eamr, degr, NR, 18);
  {
    long long total = (long long)EtotRP * 32;
    pack_ea_kernel<<<divup_ll(total, 256), 256, 0, stream>>>(rea, eamr, eahr, zbuf, ER, EtotR, total, 18);
  }

  // ---- rule path ----
  GAT(rxh, 32, rei, eahr, Wl1h, g_[1][1], Wr1h, g_[1][3], We1h, g_[1][5], g_[1][6],
      xl1h, xr1h, logr, pr, mr, denr, h1, NR, ER, 32, 100, 112, 1);
  PACK(h1, h1h, NR, 100, NR, 128);
  GAT(h1h, 128, rei, eahr, Wl2h, g_[2][1], Wr2h, g_[2][3], We2h, g_[2][5], g_[2][6],
      xl2h, xr2h, logr, pr, mr, denr, h2, NR, ER, 128, 200, 208, 1);
  PACK(h2, h2h, NR, 200, NR, 224);
  GEMM(h2h, 224, linWh, 224, lin_b, h3, nullptr, 0, NR, 400, 400, 224, 0);
  FILL(psum, 0.f, (long long)R * 400);
  FILL(pcnt, 0.f, R);
  pool_seg_kernel<<<divup_ll((long long)NR * 32, 256), 256, 0, stream>>>(h3, rbat, psum, pcnt, NR, 400);
  pool_div_kernel<<<divup_ll((long long)R * 400, 256), 256, 0, stream>>>(psum, pcnt, R, 400);
  PACK(psum, psumh, R, 400, R, 416);
  FILLH(y1h, (long long)R * 224);
  FILLH(y2h, (long long)R * 128);
  GEMM(psumh, 416, f1Wh, 416, f1_b, nullptr, y1h, 224, R, 200, 208, 416, 1);
  GEMM(y1h, 224, f2Wh, 224, f2_b, nullptr, y2h, 128, R, 100, 112, 224, 1);
  GEMM(y2h, 128, f3Wh, 128, f3_b, y3, nullptr, 0, R, 100, 112, 128, 0);

  // ---- data graph prep ----
  FILL(deg0, 0.f, N);
  FILL(eam0, 0.f, (long long)N * 18);
  deg_attr_kernel<<<divup_ll((long long)E * 32, 256), 256, 0, stream>>>(ei, ea, deg0, eam0, E, 18);
  mean_attr_kernel<<<divup_ll((long long)N * 32, 256), 256, 0, stream>>>(eam0, deg0, N, 18);
  {
    long long total = (long long)Etot0P * 32;
    pack_ea_kernel<<<divup_ll(total, 256), 256, 0, stream>>>(ea, eam0, eah0, zbuf, E, Etot0, total, 18);
  }

  // ---- data path ----
  GAT(xh, 32, ei, eah0, Wl0h, g_[0][1], Wr0h, g_[0][3], We0h, g_[0][5], g_[0][6],
      xl0h, xr0h, log0, p0, m0, den0, gout, N, E, 32, 100, 112, 0);
  FILL(xp, 0.f, 128);
  pool_all_kernel<<<256, 256, 0, stream>>>(gout, xp, N, 100);
  scale_kernel<<<1, 128, 0, stream>>>(xp, 1.0f / (float)N, 100);

  // ---- bilinear head ----
  bil_w_kernel<<<1, 128, 0, stream>>>(xp, bilW, wvec, 100);
  bil_out_kernel<<<divup_ll(R, 256), 256, 0, stream>>>(wvec, y3, bilb, (float*)d_out, R, 100);
}
